// GenerativeTransformer_88021059764450
// MI455X (gfx1250) — matrix operand order repaired and checked by reading, not compiled, not run on hardware
//
#include <hip/hip_runtime.h>
#include <math.h>

// ---------------------------------------------------------------------------
// GPT-2 forward (B=4, T=1024, E=768, L=12, H=12, V=50257) for gfx1250.
// - All matmuls on v_wmma_f32_16x16x32_bf16 (fp32 accum).
// - Weights one-shot transposed+converted to bf16 [N,K] so every GEMM's LDS
//   tiles are contiguous for both A and B fragments.
// - GEMM tiles staged by the Tensor Data Mover (tensor_load_to_lds +
//   s_wait_tensorcnt), double-buffered; cooperative-load fallback keeps the
//   same LDS layout if the builtin is unavailable.
// ---------------------------------------------------------------------------

typedef __attribute__((ext_vector_type(16))) __bf16 v16bf;
typedef __attribute__((ext_vector_type(8)))  __bf16 v8bf;
typedef __attribute__((ext_vector_type(8)))  float  v8f;
typedef __attribute__((ext_vector_type(4)))  unsigned int u32x4;
typedef __attribute__((ext_vector_type(8)))  int i32x8;
typedef __attribute__((ext_vector_type(4)))  int i32x4;

#define WMMA_BF16(a, b, c) \
  __builtin_amdgcn_wmma_f32_16x16x32_bf16(false, (a), false, (b), (short)0, (c), false, false)

#if defined(__AMDGCN__) && __has_builtin(__builtin_amdgcn_tensor_load_to_lds) && \
    __has_builtin(__builtin_amdgcn_s_wait_tensorcnt)
#define USE_TDM 1
#else
#define USE_TDM 0
#endif

constexpr int Bn = 4, Tn = 1024, En = 768, Ln = 12, Hn = 12, Vn = 50257;
constexpr int Mn = Bn * Tn;  // 4096 activation rows
constexpr float LN_EPS = 1e-5f;

enum { EPI_BIAS = 0, EPI_RES = 1, EPI_GELU = 2, EPI_NONE = 3 };

// ---------------------------------------------------------------------------
// TDM 2D tile load: global (bf16, row stride `stride0` elems) -> LDS,
// tile = tile1 rows x tile0 contiguous elems; tensor dims bound OOB (zero).
// ---------------------------------------------------------------------------
#if USE_TDM
__device__ __forceinline__ void tdm_load_2d(unsigned lds_addr, const void* gptr,
                                            unsigned tile0, unsigned tile1,
                                            unsigned long long stride0,
                                            unsigned td0, unsigned td1) {
  unsigned long long ga = (unsigned long long)gptr;
  u32x4 g0;
  g0[0] = 1u;  // count=1, user mode
  g0[1] = lds_addr;
  g0[2] = (unsigned)(ga & 0xffffffffu);
  g0[3] = (unsigned)((ga >> 32) & 0x01ffffffu) | 0x80000000u;  // type=2 ("image")
  i32x8 g1;
  g1[0] = (int)(1u << 16);  // data_size = 1 -> 2 bytes/elem; no multicast/pad
  g1[1] = (int)((td0 & 0xffffu) << 16);
  g1[2] = (int)((td0 >> 16) | ((td1 & 0xffffu) << 16));
  g1[3] = (int)((td1 >> 16) | (tile0 << 16));
  g1[4] = (int)tile1;  // tile_dim1; tile_dim2 = 0
  g1[5] = (int)(unsigned)(stride0 & 0xffffffffull);
  g1[6] = (int)(unsigned)((stride0 >> 32) & 0xffffull);
  g1[7] = 0;
  i32x4 gz = {0, 0, 0, 0};
#if __clang_major__ >= 23
  i32x8 gz8 = {0, 0, 0, 0, 0, 0, 0, 0};
  __builtin_amdgcn_tensor_load_to_lds(g0, g1, gz, gz, gz8, 0);
#else
  __builtin_amdgcn_tensor_load_to_lds(g0, g1, gz, gz, 0);
#endif
}
#endif

// ---------------------------------------------------------------------------
// fp32 -> bf16 elementwise (wte)
// ---------------------------------------------------------------------------
__global__ __launch_bounds__(256) void cvt_bf16_kernel(const float* __restrict__ in,
                                                       __bf16* __restrict__ out,
                                                       long long n) {
  long long i = (long long)blockIdx.x * blockDim.x + threadIdx.x;
  const long long stride = (long long)gridDim.x * blockDim.x;
  for (; i < n; i += stride) out[i] = (__bf16)in[i];
}

// ---------------------------------------------------------------------------
// Tiled transpose + fp32->bf16: in [L][K][N] fp32 -> out [L][N][K] bf16.
// K, N multiples of 32. block (32,8), grid (N/32, K/32, L).
// ---------------------------------------------------------------------------
__global__ __launch_bounds__(256) void transpose_cvt_kernel(const float* __restrict__ in,
                                                            __bf16* __restrict__ out,
                                                            int K, int N) {
  __shared__ float tile[32][33];
  const int l = blockIdx.z;
  const float* ip = in + (size_t)l * K * N;
  __bf16* op = out + (size_t)l * N * K;
  const int nb = blockIdx.x * 32, kb = blockIdx.y * 32;
  const int tx = threadIdx.x, ty = threadIdx.y;
#pragma unroll
  for (int j = 0; j < 32; j += 8)
    tile[ty + j][tx] = ip[(size_t)(kb + ty + j) * N + nb + tx];
  __syncthreads();
#pragma unroll
  for (int j = 0; j < 32; j += 8)
    op[(size_t)(nb + ty + j) * K + kb + tx] = (__bf16)tile[tx][ty + j];
}

// ---------------------------------------------------------------------------
// Embedding: x[b*T+t][:] = wte[idx[row]][:] + wpe[t][:]   (fp32 residual)
// ---------------------------------------------------------------------------
__global__ __launch_bounds__(256) void embed_kernel(const int* __restrict__ idx,
                                                    const float* __restrict__ wte,
                                                    const float* __restrict__ wpe,
                                                    float* __restrict__ x) {
  const int row = blockIdx.x;
  const int t = row & (Tn - 1);
  const int tok = idx[row];
  const float* a = wte + (size_t)tok * En;
  const float* p = wpe + (size_t)t * En;
  float* xp = x + (size_t)row * En;
  for (int e = threadIdx.x; e < En; e += 256) xp[e] = a[e] + p[e];
}

// ---------------------------------------------------------------------------
// LayerNorm (fp32 in, bf16 out): one block per row
// ---------------------------------------------------------------------------
__global__ __launch_bounds__(256) void ln_kernel(const float* __restrict__ x,
                                                 const float* __restrict__ w,
                                                 const float* __restrict__ b,
                                                 __bf16* __restrict__ out) {
  __shared__ float s1[256];
  __shared__ float s2[256];
  const int row = blockIdx.x;
  const int tid = threadIdx.x;
  const float* xp = x + (size_t)row * En;
  float sum = 0.f, sq = 0.f;
  for (int e = tid; e < En; e += 256) {
    float v = xp[e];
    sum += v;
    sq += v * v;
  }
  s1[tid] = sum;
  s2[tid] = sq;
  __syncthreads();
  for (int o = 128; o > 0; o >>= 1) {
    if (tid < o) {
      s1[tid] += s1[tid + o];
      s2[tid] += s2[tid + o];
    }
    __syncthreads();
  }
  const float mean = s1[0] * (1.0f / En);
  const float var = s2[0] * (1.0f / En) - mean * mean;
  const float rs = rsqrtf(var + LN_EPS);
  __bf16* op = out + (size_t)row * En;
  for (int e = tid; e < En; e += 256) op[e] = (__bf16)((xp[e] - mean) * rs * w[e] + b[e]);
}

// ---------------------------------------------------------------------------
// bf16 WMMA GEMM: out[M,N] = A[M,K] * W^T (+bias, +residual, +GELU)
// W stored [N,K] row-major (pre-transposed). Block = 256 threads (8 waves),
// C tile 128(M) x 64(N), K step 64, double-buffered LDS; both LDS tiles are
// k-contiguous so all fragments are 16B vector LDS reads.
// ---------------------------------------------------------------------------
template <int EPI, bool OUTBF>
__global__ __launch_bounds__(256) void gemm_kernel(const __bf16* __restrict__ A,
                                                   const __bf16* __restrict__ W,
                                                   const float* __restrict__ bias,
                                                   const float* __restrict__ res,
                                                   void* __restrict__ outp,
                                                   int M, int N, int K) {
  __shared__ __bf16 As[2][128][64];  // [m][k]
  __shared__ __bf16 Ws[2][64][64];   // [n][k]

  const int tid = threadIdx.x;
  const int lane = tid & 31;
  const int wv = tid >> 5;
  const int n0 = blockIdx.x * 64;
  const int m0 = blockIdx.y * 128;

  const int mrow = lane & 15;       // A row / B column within 16
  const int ka0 = (lane >> 4) * 8;  // A frag K runs [ka0,ka0+8) and [ka0+16,ka0+24)
  const int kb0 = (lane >> 4) * 16; // unused; B frag K runs use ka0, same as A

  v8f acc0 = {}, acc1 = {}, acc2 = {}, acc3 = {};

  const int S = K / 64;

  auto stage = [&](int buf, int kt) {
#if USE_TDM
    if (wv == 0) {
      tdm_load_2d((unsigned)(uintptr_t)&As[buf][0][0], A + (size_t)m0 * K + kt,
                  64u, 128u, (unsigned long long)K, (unsigned)(K - kt), (unsigned)(M - m0));
      tdm_load_2d((unsigned)(uintptr_t)&Ws[buf][0][0], W + (size_t)n0 * K + kt,
                  64u, 64u, (unsigned long long)K, (unsigned)(K - kt), (unsigned)(N - n0));
    }
#else
    {  // cooperative loads, same LDS layouts as the TDM path
      const int r = tid >> 1;            // 0..127
      const int c = (tid & 1) * 32;      // 0,32
      const __bf16* ap = A + (size_t)(m0 + r) * K + kt + c;
#pragma unroll
      for (int i = 0; i < 32; i += 8) *(v8bf*)&As[buf][r][c + i] = *(const v8bf*)(ap + i);
    }
    {
      const int nn = tid >> 2;           // 0..63
      const int c = (tid & 3) * 16;      // k offset
      const bool ok = (n0 + nn) < N;
      const __bf16* wp = W + (size_t)(n0 + nn) * K + kt + c;
#pragma unroll
      for (int i = 0; i < 16; ++i) Ws[buf][nn][c + i] = ok ? wp[i] : (__bf16)0.f;
    }
#endif
  };

  auto wait_stage = [&]() {
#if USE_TDM
    if (wv == 0) __builtin_amdgcn_s_wait_tensorcnt(0);
#endif
    __syncthreads();
  };

  stage(0, 0);
  for (int s = 0; s < S; ++s) {
    wait_stage();  // buffer (s&1) ready; also fences previous compute
    if (s + 1 < S) stage((s + 1) & 1, (s + 1) * 64);
    const int buf = s & 1;
#pragma unroll
    for (int c = 0; c < 2; ++c) {
      v16bf a;
      {
        const __bf16* ap = &As[buf][wv * 16 + mrow][c * 32];
        v8bf lo = *(const v8bf*)(ap + ka0);
        v8bf hi = *(const v8bf*)(ap + ka0 + 16);
#pragma unroll
        for (int i = 0; i < 8; ++i) {
          a[i] = lo[i];
          a[8 + i] = hi[i];
        }
      }
#pragma unroll
      for (int st = 0; st < 4; ++st) {
        const __bf16* bp = &Ws[buf][st * 16 + mrow][c * 32];
        v8bf b0 = *(const v8bf*)(bp + ka0);
        v8bf b1 = *(const v8bf*)(bp + ka0 + 16);
        v16bf bfrag;
#pragma unroll
        for (int i = 0; i < 8; ++i) {
          bfrag[i] = b0[i];
          bfrag[8 + i] = b1[i];
        }
        if (st == 0) acc0 = WMMA_BF16(a, bfrag, acc0);
        else if (st == 1) acc1 = WMMA_BF16(a, bfrag, acc1);
        else if (st == 2) acc2 = WMMA_BF16(a, bfrag, acc2);
        else acc3 = WMMA_BF16(a, bfrag, acc3);
      }
    }
  }

  // epilogue: C layout VGPR r -> row (r + 8*(lane>>4)), col = lane&15
  const int hi2 = lane >> 4;
  const int col = lane & 15;
  v8f accv[4] = {acc0, acc1, acc2, acc3};
#pragma unroll
  for (int st = 0; st < 4; ++st) {
    const int n = n0 + st * 16 + col;
    if (n >= N) continue;
    const float bv = (EPI == EPI_NONE) ? 0.f : bias[n];
#pragma unroll
    for (int r = 0; r < 8; ++r) {
      const int m = m0 + wv * 16 + hi2 * 8 + r;
      float v = accv[st][r] + bv;
      if (EPI == EPI_GELU) v = 0.5f * v * (1.0f + erff(v * 0.70710678118654752f));
      if (EPI == EPI_RES) v += res[(size_t)m * N + n];
      if (OUTBF) ((__bf16*)outp)[(size_t)m * N + n] = (__bf16)v;
      else ((float*)outp)[(size_t)m * N + n] = v;
    }
  }
}

// ---------------------------------------------------------------------------
// Causal attention (bf16 qkv in, bf16 y out). One wave per (b, h, 16-q tile).
// QK^T via WMMA -> LDS fp32 scores -> softmax -> in-place bf16 probs -> P*V.
// V tiles staged through LDS with vector loads (no scalar global gathers).
// ---------------------------------------------------------------------------
__global__ __launch_bounds__(32) void attn_kernel(const __bf16* __restrict__ qkv,
                                                  __bf16* __restrict__ y) {
  __shared__ float sc[16][Tn];      // 64 KB scores -> probs
  __shared__ __bf16 vt[32][64];     // 4 KB staged V tile (32 keys x 64 d)

  const int lane = threadIdx.x;
  const int bid = blockIdx.x;
  const int qt = bid & 63;
  const int h = (bid >> 6) % Hn;
  const int b = bid / (64 * Hn);

  const int mrow = lane & 15;
  const int hi = lane >> 4;
  const int ka0 = hi * 8;
  const int kb0 = hi * 16;
  const float scale = 0.125f;  // 1/sqrt(64)

  // Q fragments (2 chunks of d=32)
  v16bf qf0, qf1;
  {
    const int tq = qt * 16 + mrow;
    const __bf16* qp = qkv + (size_t)(b * Tn + tq) * (3 * En) + h * 64;
    v8bf l0 = *(const v8bf*)(qp + ka0);
    v8bf h0 = *(const v8bf*)(qp + ka0 + 16);
    v8bf l1 = *(const v8bf*)(qp + 32 + ka0);
    v8bf h1 = *(const v8bf*)(qp + 32 + ka0 + 16);
#pragma unroll
    for (int i = 0; i < 8; ++i) {
      qf0[i] = l0[i];
      qf0[8 + i] = h0[i];
      qf1[i] = l1[i];
      qf1[8 + i] = h1[i];
    }
  }

  // scores
  for (int kt = 0; kt <= qt; ++kt) {
    const int tk = kt * 16 + mrow;
    const __bf16* kp = qkv + (size_t)(b * Tn + tk) * (3 * En) + En + h * 64;
    v8f acc = {};
#pragma unroll
    for (int c = 0; c < 2; ++c) {
      v8bf b0 = *(const v8bf*)(kp + c * 32 + ka0);
      v8bf b1 = *(const v8bf*)(kp + c * 32 + ka0 + 16);
      v16bf kf;
#pragma unroll
      for (int i = 0; i < 8; ++i) {
        kf[i] = b0[i];
        kf[8 + i] = b1[i];
      }
      acc = WMMA_BF16(c == 0 ? qf0 : qf1, kf, acc);
    }
#pragma unroll
    for (int r = 0; r < 8; ++r) {
      const int Mq = r + 8 * hi;
      const int tq = qt * 16 + Mq;
      const int tkk = kt * 16 + mrow;
      sc[Mq][kt * 16 + mrow] = (tkk <= tq) ? acc[r] * scale : -3.0e38f;
    }
  }
  __syncthreads();

  // softmax per row; in-place bf16 prob conversion (write@2j never clobbers unread read@4j)
  const int Lk = (qt + 1) * 16;
  const int Lpad = (Lk + 31) & ~31;
  if (lane < 16) {
    float* row = &sc[lane][0];
    __bf16* prow = (__bf16*)row;
    float mx = -3.4e38f;
    for (int j = 0; j < Lk; ++j) mx = fmaxf(mx, row[j]);
    float sum = 0.f;
    for (int j = 0; j < Lk; ++j) {
      float e = expf(row[j] - mx);
      row[j] = e;
      sum += e;
    }
    const float inv = 1.0f / sum;
    for (int j = 0; j < Lk; ++j) prow[j] = (__bf16)(row[j] * inv);
    for (int j = Lk; j < Lpad; ++j) prow[j] = (__bf16)0.f;
  }
  __syncthreads();

  // P * V, V staged per 32-key chunk into LDS with vector loads
  v8f ya0 = {}, ya1 = {}, ya2 = {}, ya3 = {};
  const __bf16* prow = (const __bf16*)&sc[mrow][0];
  for (int k2 = 0; k2 < Lpad; k2 += 32) {
    {
      const __bf16* vp = qkv + (size_t)(b * Tn + k2 + lane) * (3 * En) + 2 * En + h * 64;
#pragma unroll
      for (int i = 0; i < 64; i += 8) *(v8bf*)&vt[lane][i] = *(const v8bf*)(vp + i);
    }
    __syncthreads();

    v16bf pf;
#pragma unroll
    for (int i = 0; i < 8; ++i) {
      pf[i] = prow[k2 + kb0 + i];
      pf[8 + i] = prow[k2 + kb0 + 8 + i];
    }
#pragma unroll
    for (int st = 0; st < 4; ++st) {
      v16bf vf;
#pragma unroll
      for (int i = 0; i < 16; ++i) vf[i] = vt[kb0 + i][st * 16 + mrow];
      if (st == 0) ya0 = WMMA_BF16(pf, vf, ya0);
      else if (st == 1) ya1 = WMMA_BF16(pf, vf, ya1);
      else if (st == 2) ya2 = WMMA_BF16(pf, vf, ya2);
      else ya3 = WMMA_BF16(pf, vf, ya3);
    }
    __syncthreads();
  }

#pragma unroll
  for (int r = 0; r < 8; ++r) {
    const int Mq = r + 8 * hi;
    const int tq = qt * 16 + Mq;
    __bf16* yp = y + (size_t)(b * Tn + tq) * En + h * 64;
    yp[mrow] = (__bf16)ya0[r];
    yp[16 + mrow] = (__bf16)ya1[r];
    yp[32 + mrow] = (__bf16)ya2[r];
    yp[48 + mrow] = (__bf16)ya3[r];
  }
}

// ---------------------------------------------------------------------------
// Host-side dispatch
// ---------------------------------------------------------------------------
static void launch_gemm(int epi, const __bf16* A, const __bf16* W, const float* bias,
                        const float* res, void* out, int M, int N, int K,
                        hipStream_t stream) {
  dim3 grid((N + 63) / 64, M / 128), blk(256);
  if (epi == EPI_BIAS)
    gemm_kernel<EPI_BIAS, true><<<grid, blk, 0, stream>>>(A, W, bias, res, out, M, N, K);
  else if (epi == EPI_RES)
    gemm_kernel<EPI_RES, false><<<grid, blk, 0, stream>>>(A, W, bias, res, out, M, N, K);
  else if (epi == EPI_GELU)
    gemm_kernel<EPI_GELU, true><<<grid, blk, 0, stream>>>(A, W, bias, res, out, M, N, K);
  else
    gemm_kernel<EPI_NONE, false><<<grid, blk, 0, stream>>>(A, W, bias, res, out, M, N, K);
}

static void transpose_cvt(const float* in, __bf16* out, int L, int K, int N,
                          hipStream_t s) {
  dim3 blk(32, 8), grid(N / 32, K / 32, L);
  transpose_cvt_kernel<<<grid, blk, 0, s>>>(in, out, K, N);
}

extern "C" void kernel_launch(void* const* d_in, const int* in_sizes, int n_in,
                              void* d_out, int out_size, void* d_ws, size_t ws_size,
                              hipStream_t stream) {
  const int* x_idx = (const int*)d_in[0];
  const float* wte = (const float*)d_in[1];
  const float* wpe = (const float*)d_in[2];
  const float* ln1_w = (const float*)d_in[3];
  const float* ln1_b = (const float*)d_in[4];
  const float* attn_w = (const float*)d_in[5];
  const float* attn_b = (const float*)d_in[6];
  const float* proj_w = (const float*)d_in[7];
  const float* proj_b = (const float*)d_in[8];
  const float* ln2_w = (const float*)d_in[9];
  const float* ln2_b = (const float*)d_in[10];
  const float* fc1_w = (const float*)d_in[11];
  const float* fc1_b = (const float*)d_in[12];
  const float* fc2_w = (const float*)d_in[13];
  const float* fc2_b = (const float*)d_in[14];
  const float* lnf_w = (const float*)d_in[15];
  const float* lnf_b = (const float*)d_in[16];

  // ---- workspace layout ----
  float* x = (float*)d_ws;                       // [4096,768] fp32 residual
  __bf16* bp = (__bf16*)(x + (size_t)Mn * En);
  __bf16* hb = bp;   bp += (size_t)Mn * En;          // LN out
  __bf16* qkvb = bp; bp += (size_t)Mn * 3 * En;      // qkv
  __bf16* yb = bp;   bp += (size_t)Mn * En;          // attn out
  __bf16* ffb = bp;  bp += (size_t)Mn * 4 * En;      // fc1+gelu out
  __bf16* wteb = bp; bp += (size_t)Vn * En;          // bf16 weights, all [N,K]
  __bf16* awT = bp;  bp += (size_t)Ln * 3 * En * En; // [L][3E][E]
  __bf16* pwT = bp;  bp += (size_t)Ln * En * En;     // [L][E][E]
  __bf16* f1T = bp;  bp += (size_t)Ln * 4 * En * En; // [L][4E][E]
  __bf16* f2T = bp;  bp += (size_t)Ln * En * 4 * En; // [L][E][4E]

  // one-shot weight conversion/transpose (deterministic each call)
  cvt_bf16_kernel<<<2048, 256, 0, stream>>>(wte, wteb, (long long)Vn * En);
  transpose_cvt(attn_w, awT, Ln, En, 3 * En, stream);
  transpose_cvt(proj_w, pwT, Ln, En, En, stream);
  transpose_cvt(fc1_w, f1T, Ln, En, 4 * En, stream);
  transpose_cvt(fc2_w, f2T, Ln, 4 * En, En, stream);

  embed_kernel<<<Mn, 256, 0, stream>>>(x_idx, wte, wpe, x);

  for (int l = 0; l < Ln; ++l) {
    ln_kernel<<<Mn, 256, 0, stream>>>(x, ln1_w + (size_t)l * En, ln1_b + (size_t)l * En, hb);
    launch_gemm(EPI_BIAS, hb, awT + (size_t)l * 3 * En * En, attn_b + (size_t)l * 3 * En,
                nullptr, qkvb, Mn, 3 * En, En, stream);
    attn_kernel<<<Bn * Hn * (Tn / 16), 32, 0, stream>>>(qkvb, yb);
    launch_gemm(EPI_RES, yb, pwT + (size_t)l * En * En, proj_b + (size_t)l * En,
                x, x, Mn, En, En, stream);
    ln_kernel<<<Mn, 256, 0, stream>>>(x, ln2_w + (size_t)l * En, ln2_b + (size_t)l * En, hb);
    launch_gemm(EPI_GELU, hb, f1T + (size_t)l * 4 * En * En, fc1_b + (size_t)l * 4 * En,
                nullptr, ffb, Mn, 4 * En, En, stream);
    launch_gemm(EPI_RES, ffb, f2T + (size_t)l * En * 4 * En, fc2_b + (size_t)l * En,
                x, x, Mn, En, 4 * En, stream);
  }

  ln_kernel<<<Mn, 256, 0, stream>>>(x, lnf_w, lnf_b, hb);
  // logits = h @ wte^T  (wteb is [V, E] row-major = [N,K]; D# OOB zero-fills V edge)
  launch_gemm(EPI_NONE, hb, wteb, nullptr, nullptr, (float*)d_out, Mn, Vn, En, stream);
}
